// MoE_11991548691210
// MI455X (gfx1250) — matrix operand order repaired and checked by reading, not compiled, not run on hardware
//
#include <hip/hip_runtime.h>
#include <hip/hip_bf16.h>
#include <math.h>

// ---------------- problem constants ----------------
#define TOKENS   4096
#define DIM      1024
#define HID      2816
#define NEXP     8
#define TOPK     2
#define TR       (TOKENS * TOPK)      // 8192 routed rows
#define ROWS_MAX 8320                 // TR + NEXP*16 padding headroom
#define MTILES   (ROWS_MAX / 16)      // 520

// ---------------- workspace layout (bytes) ----------------
#define SEL_E_OFF  0                                       // int[TR]
#define SEL_S_OFF  (TR * 4)                                // float[TR]
#define RBASE_OFF  (TR * 8)                                // int[NEXP+1]
#define CNT_OFF    (RBASE_OFF + 64)                        // int[NEXP]
#define ROWSRC_OFF (CNT_OFF + 64)                          // int[ROWS_MAX]
#define XR_OFF     ((size_t)131072)                        // bf16[ROWS_MAX*DIM]
#define HB_OFF     (XR_OFF + (size_t)ROWS_MAX * DIM * 2)   // bf16[ROWS_MAX*HID]
#define W1T_OFF    (HB_OFF + (size_t)ROWS_MAX * HID * 2)   // bf16[E][HID][DIM]
#define W3T_OFF    (W1T_OFF + (size_t)NEXP * DIM * HID * 2)
#define W2T_OFF    (W3T_OFF + (size_t)NEXP * DIM * HID * 2) // bf16[E][DIM][HID]

typedef __bf16 bf16_t;
typedef bf16_t v16bf __attribute__((ext_vector_type(16)));
typedef float  v8f   __attribute__((ext_vector_type(8)));

union ATile { v16bf v; unsigned int u[8]; };

__device__ __forceinline__ unsigned short f2bf(float f) {
  unsigned int u = __float_as_uint(f);
  u += 0x7FFFu + ((u >> 16) & 1u);          // round-to-nearest-even
  return (unsigned short)(u >> 16);
}

// A operand: 16x32 bf16 tile from row-major bf16 (ld elems/row).
// lanes 0-15 -> M=lane, per-lane K in {hi*8+0..7, 16+hi*8+0..7} as bf16 pairs.
__device__ __forceinline__ void load_a_bf16(ATile& a, const unsigned short* __restrict__ buf,
                                            int row0, int ld, int k0, int lane) {
  int m = lane & 15, hi = lane >> 4;
  const unsigned short* p = buf + (size_t)(row0 + m) * ld + k0 + hi * 8;
#pragma unroll
  for (int v = 0; v < 4; ++v) a.u[v]     = *(const unsigned int*)(p + 2 * v);
#pragma unroll
  for (int v = 0; v < 4; ++v) a.u[4 + v] = *(const unsigned int*)(p + 16 + 2 * v);
  __builtin_prefetch(p + 32, 0, 1);          // next K-step (global_prefetch_b8)
}

// B operand: 32x16 tile from pre-transposed bf16 weights WT[N][K] (ld = K).
// lanes 0-15 -> N=lane, per-lane K in {hi*8+0..7, 16+hi*8+0..7} (matches A).
__device__ __forceinline__ void load_b_bf16(ATile& b, const unsigned short* __restrict__ buf,
                                            int n0, int ld, int k0, int lane) {
  int n = lane & 15, kh = lane >> 4;
  const unsigned short* p = buf + (size_t)(n0 + n) * ld + k0 + kh * 8;
#pragma unroll
  for (int v = 0; v < 4; ++v) b.u[v]     = *(const unsigned int*)(p + 2 * v);
#pragma unroll
  for (int v = 0; v < 4; ++v) b.u[4 + v] = *(const unsigned int*)(p + 16 + 2 * v);
  __builtin_prefetch(p + 32, 0, 1);          // next K-step
}

// ---------------- kernel 0: zero output ----------------
__global__ __launch_bounds__(256) void moe_zero(float* __restrict__ out) {
  int base = blockIdx.x * 1024 + threadIdx.x;
#pragma unroll
  for (int j = 0; j < 4; ++j) out[base + j * 256] = 0.0f;
}

// ---------------- kernel W: fp32 [E][K][N] -> bf16 [E][N][K] (LDS transpose) ----
__global__ __launch_bounds__(256) void moe_cvt_w(const float* __restrict__ W,
                                                 unsigned short* __restrict__ WT,
                                                 int K, int N) {
  __shared__ float tile[32][33];
  const float*    Wp = W  + (size_t)blockIdx.z * K * N;
  unsigned short* Tp = WT + (size_t)blockIdx.z * K * N;
  int n0 = blockIdx.x * 32, k0 = blockIdx.y * 32;
  int tx = threadIdx.x & 31, ty = threadIdx.x >> 5;
#pragma unroll
  for (int j = 0; j < 4; ++j) {
    int r = ty + j * 8;
    tile[r][tx] = Wp[(size_t)(k0 + r) * N + n0 + tx];
  }
  __syncthreads();
#pragma unroll
  for (int j = 0; j < 4; ++j) {
    int r = ty + j * 8;
    Tp[(size_t)(n0 + r) * K + k0 + tx] = f2bf(tile[tx][r]);
  }
}

// ---------------- kernel 1: routing (softmax + top-2) ----------------
__global__ __launch_bounds__(128) void moe_route(const float* __restrict__ x,
                                                 const float* __restrict__ gw,
                                                 int* __restrict__ sel_e,
                                                 float* __restrict__ sel_s) {
  __shared__ float sg[NEXP * DIM];            // 32 KB of the 320 KB LDS
  for (int i = threadIdx.x; i < NEXP * DIM; i += 128) sg[i] = gw[i];
  __syncthreads();

  int t = blockIdx.x * 128 + threadIdx.x;
  const float* xr = x + (size_t)t * DIM;
  float acc[NEXP];
#pragma unroll
  for (int e = 0; e < NEXP; ++e) acc[e] = 0.0f;
  for (int d = 0; d < DIM; ++d) {
    float xv = xr[d];
#pragma unroll
    for (int e = 0; e < NEXP; ++e) acc[e] += xv * sg[e * DIM + d];
  }
  float m = acc[0];
#pragma unroll
  for (int e = 1; e < NEXP; ++e) m = fmaxf(m, acc[e]);
  float sum = 0.0f, p[NEXP];
#pragma unroll
  for (int e = 0; e < NEXP; ++e) { p[e] = __expf(acc[e] - m); sum += p[e]; }
  float inv = 1.0f / sum;
#pragma unroll
  for (int e = 0; e < NEXP; ++e) p[e] *= inv;
  int e0 = 0;
#pragma unroll
  for (int e = 1; e < NEXP; ++e) if (p[e] > p[e0]) e0 = e;
  int e1 = (e0 == 0) ? 1 : 0;
#pragma unroll
  for (int e = 0; e < NEXP; ++e) if (e != e0 && p[e] > p[e1]) e1 = e;
  sel_e[t * 2 + 0] = e0;  sel_s[t * 2 + 0] = p[e0];
  sel_e[t * 2 + 1] = e1;  sel_s[t * 2 + 1] = p[e1];
}

// ---------------- kernel 2: deterministic per-expert compaction ----------------
__global__ __launch_bounds__(256) void moe_compact(const int* __restrict__ sel_e,
                                                   int* __restrict__ cnt,
                                                   int* __restrict__ rbase,
                                                   int* __restrict__ row_src) {
  int tid = threadIdx.x;
  if (tid < NEXP) {
    int c = 0;
    for (int i = 0; i < TR; ++i) c += (sel_e[i] == tid);
    cnt[tid] = c;
  }
  __syncthreads();
  if (tid == 0) {
    int off = 0;
    for (int e = 0; e < NEXP; ++e) {
      rbase[e] = off;
      off += (cnt[e] + 15) & ~15;             // pad each bucket to 16 rows
    }
    rbase[NEXP] = off;                        // padded total
  }
  __syncthreads();
  if (tid < NEXP) {
    int pos = rbase[tid];
    for (int i = 0; i < TR; ++i)
      if (sel_e[i] == tid) row_src[pos++] = i;   // stable (token order)
    int end = rbase[tid] + ((cnt[tid] + 15) & ~15);
    for (; pos < end; ++pos) row_src[pos] = -1;  // padding rows
  }
}

// ---------------- kernel 3: gather + pre-scale -> bf16 Xr ----------------
__global__ __launch_bounds__(256) void moe_gather(const float* __restrict__ x,
                                                  const float* __restrict__ sel_s,
                                                  const int* __restrict__ row_src,
                                                  const int* __restrict__ rbase,
                                                  unsigned short* __restrict__ Xr) {
  int r = blockIdx.x;
  if (r >= rbase[NEXP]) return;
  int src = row_src[r];
  if (src < 0) {
#pragma unroll
    for (int j = 0; j < 4; ++j) Xr[(size_t)r * DIM + threadIdx.x + j * 256] = 0;
    return;
  }
  int tok = src >> 1;
  float s = sel_s[src];
#pragma unroll
  for (int j = 0; j < 4; ++j) {
    int c = threadIdx.x + j * 256;
    Xr[(size_t)r * DIM + c] = f2bf(x[(size_t)tok * DIM + c] * s);
  }
}

// ---------------- kernel 4: h = silu(Xr@w1) * (Xr@w3)  (all-bf16 WMMA) --------
__global__ __launch_bounds__(256) void moe_ffn1(const unsigned short* __restrict__ Xr,
                                                const unsigned short* __restrict__ w1t,
                                                const unsigned short* __restrict__ w3t,
                                                const int* __restrict__ rbase,
                                                unsigned short* __restrict__ Hb) {
  int m0 = blockIdx.x * 16;
  if (m0 >= rbase[NEXP]) return;              // uniform per workgroup
  int e = 0;
#pragma unroll
  for (int i = 0; i < NEXP - 1; ++i) if (m0 >= rbase[i + 1]) e = i + 1;

  int lane = threadIdx.x & 31;
  int n0 = blockIdx.y * 128 + (threadIdx.x >> 5) * 16;   // 8 waves x 16 cols
  const unsigned short* W1 = w1t + (size_t)e * DIM * HID;  // [HID][DIM] bf16
  const unsigned short* W3 = w3t + (size_t)e * DIM * HID;

  v8f acc1 = {}; v8f acc3 = {};
  for (int k0 = 0; k0 < DIM; k0 += 32) {
    ATile a, b1, b3;
    load_a_bf16(a, Xr, m0, DIM, k0, lane);
    load_b_bf16(b1, W1, n0, DIM, k0, lane);
    load_b_bf16(b3, W3, n0, DIM, k0, lane);
    acc1 = __builtin_amdgcn_wmma_f32_16x16x32_bf16(false, a.v, false, b1.v,
                                                   (short)0, acc1, false, false);
    acc3 = __builtin_amdgcn_wmma_f32_16x16x32_bf16(false, a.v, false, b3.v,
                                                   (short)0, acc3, false, false);
  }
  int hi = lane >> 4, n = lane & 15;
#pragma unroll
  for (int r = 0; r < 8; ++r) {               // C/D layout: M = r + 8*hi, N = lane&15
    float g = acc1[r];
    float h = (g / (1.0f + __expf(-g))) * acc3[r];
    Hb[(size_t)(m0 + r + 8 * hi) * HID + n0 + n] = f2bf(h);
  }
}

// ---------------- kernel 5: out += h @ w2, scatter-add to tokens --------------
__global__ __launch_bounds__(256) void moe_ffn2(const unsigned short* __restrict__ Hb,
                                                const unsigned short* __restrict__ w2t,
                                                const int* __restrict__ rbase,
                                                const int* __restrict__ row_src,
                                                float* __restrict__ out) {
  int m0 = blockIdx.x * 16;
  if (m0 >= rbase[NEXP]) return;
  int e = 0;
#pragma unroll
  for (int i = 0; i < NEXP - 1; ++i) if (m0 >= rbase[i + 1]) e = i + 1;

  int lane = threadIdx.x & 31;
  int n0 = blockIdx.y * 128 + (threadIdx.x >> 5) * 16;
  const unsigned short* W2 = w2t + (size_t)e * HID * DIM;  // [DIM][HID] bf16

  v8f acc = {};
  for (int k0 = 0; k0 < HID; k0 += 32) {
    ATile a, b;
    load_a_bf16(a, Hb, m0, HID, k0, lane);
    load_b_bf16(b, W2, n0, HID, k0, lane);
    acc = __builtin_amdgcn_wmma_f32_16x16x32_bf16(false, a.v, false, b.v,
                                                  (short)0, acc, false, false);
  }
  int hi = lane >> 4, n = lane & 15;
#pragma unroll
  for (int r = 0; r < 8; ++r) {
    int row = m0 + r + 8 * hi;
    int src = row_src[row];
    if (src >= 0) {
      int tok = src >> 1;
      atomicAdd(&out[(size_t)tok * DIM + n0 + n], acc[r]);
    }
  }
}

// ---------------- host launcher ----------------
extern "C" void kernel_launch(void* const* d_in, const int* in_sizes, int n_in,
                              void* d_out, int out_size, void* d_ws, size_t ws_size,
                              hipStream_t stream) {
  (void)in_sizes; (void)n_in; (void)out_size; (void)ws_size;
  const float* x      = (const float*)d_in[0];
  const float* gate_w = (const float*)d_in[1];
  const float* w1     = (const float*)d_in[2];
  const float* w2     = (const float*)d_in[3];
  const float* w3     = (const float*)d_in[4];
  float* out = (float*)d_out;

  char* ws = (char*)d_ws;
  int*            sel_e   = (int*)(ws + SEL_E_OFF);
  float*          sel_s   = (float*)(ws + SEL_S_OFF);
  int*            rbase   = (int*)(ws + RBASE_OFF);
  int*            cnt     = (int*)(ws + CNT_OFF);
  int*            row_src = (int*)(ws + ROWSRC_OFF);
  unsigned short* Xr      = (unsigned short*)(ws + XR_OFF);
  unsigned short* Hb      = (unsigned short*)(ws + HB_OFF);
  unsigned short* w1t     = (unsigned short*)(ws + W1T_OFF);
  unsigned short* w3t     = (unsigned short*)(ws + W3T_OFF);
  unsigned short* w2t     = (unsigned short*)(ws + W2T_OFF);

  moe_zero<<<TOKENS * DIM / 1024, 256, 0, stream>>>(out);
  // one-time (per launch) weight convert+transpose to WMMA-friendly bf16
  moe_cvt_w<<<dim3(HID / 32, DIM / 32, NEXP), 256, 0, stream>>>(w1, w1t, DIM, HID);
  moe_cvt_w<<<dim3(HID / 32, DIM / 32, NEXP), 256, 0, stream>>>(w3, w3t, DIM, HID);
  moe_cvt_w<<<dim3(DIM / 32, HID / 32, NEXP), 256, 0, stream>>>(w2, w2t, HID, DIM);

  moe_route<<<TOKENS / 128, 128, 0, stream>>>(x, gate_w, sel_e, sel_s);
  moe_compact<<<1, 256, 0, stream>>>(sel_e, cnt, rbase, row_src);
  moe_gather<<<ROWS_MAX, 256, 0, stream>>>(x, sel_s, row_src, rbase, Xr);
  moe_ffn1<<<dim3(MTILES, HID / 128), 256, 0, stream>>>(Xr, w1t, w3t, rbase, Hb);
  moe_ffn2<<<dim3(MTILES, DIM / 128), 256, 0, stream>>>(Hb, w2t, rbase, row_src, out);
}
